// TemporalAttention_74199855006116
// MI455X (gfx1250) — matrix operand order repaired and checked by reading, not compiled, not run on hardware
//
#include <hip/hip_runtime.h>
#include <hip/hip_bf16.h>
#include <math.h>

#define B_  8
#define S_  1024
#define E_  1024
#define H_  16
#define DH_ 64
#define M_  (B_*S_)

typedef _Float16 v16h __attribute__((ext_vector_type(16)));
typedef _Float16 h8   __attribute__((ext_vector_type(8)));
typedef float    v8f  __attribute__((ext_vector_type(8)));
typedef float    f4   __attribute__((ext_vector_type(4)));

static __device__ inline v16h pack16(h8 lo, h8 hi) {
  v16h r;
#pragma unroll
  for (int i = 0; i < 8; ++i) { r[i] = lo[i]; r[8 + i] = hi[i]; }
  return r;
}

// A-fragment (16x32 f16, MxK). lane: m = lane&15, kb = (lane<16)?0:8
// e[0..7] = row[kb..kb+7], e[8..15] = row[16+kb..16+kb+7]
static __device__ inline v16h load_afrag(const _Float16* rowp, int kb) {
  h8 lo = *(const h8*)(rowp + kb);
  h8 hi = *(const h8*)(rowp + 16 + kb);
  return pack16(lo, hi);
}

// B-fragment (32x16 f16, KxN). lane: n = lane&15, kb2 = (lane<16)?0:8
// e[0..7] = source row at [kb2 .. kb2+7], e[8..15] = [16+kb2 .. 16+kb2+7]; caller pre-offsets kb2.
static __device__ inline v16h load_bfrag(const _Float16* p) {
  h8 lo = *(const h8*)(p);
  h8 hi = *(const h8*)(p + 16);
  return pack16(lo, hi);
}

// Async global -> LDS (CDNA5 TDM-adjacent path, ASYNCcnt tracked).
// vdst = wave-relative LDS byte offset (low 32 bits of flat shared pointer).
static __device__ inline void async_ld_b128(const void* gaddr, void* lds) {
  unsigned lo = (unsigned)(size_t)lds;
  unsigned long long ga = (unsigned long long)(size_t)gaddr;
  asm volatile("global_load_async_to_lds_b128 %0, %1, off"
               :: "v"(lo), "v"(ga) : "memory");
}

// ---------------------------------------------------------------------------
// Y[M,N] = X[M,K] (f32) @ W[N,K]^T (f32) + bias[N]; f16 GEMM, f32 accumulate.
// Tiles staged via async global->LDS loads, double-buffered so stage k+1's
// copy overlaps stage k's WMMAs.
// TRANS=true stores Y as [B][E][S] (d-major) for the V operand of attention.
// ---------------------------------------------------------------------------
template <typename TOUT, bool TRANS>
__global__ __launch_bounds__(128)
void gemm_nt_bias(const float* __restrict__ X, const float* __restrict__ W,
                  const float* __restrict__ bias, TOUT* __restrict__ Y,
                  int M, int N, int K)
{
  __shared__ float lx[2][64 * 36];  // 64 rows x 32 k, padded stride 36
  __shared__ float lw[2][64 * 36];
  const int tid  = threadIdx.x;
  const int wid  = tid >> 5, lane = tid & 31;
  const int half = lane >> 4;
  const int l15  = lane & 15;
  const int n0 = blockIdx.x * 64;
  const int m0 = blockIdx.y * 64;

  v8f acc[4] = {};

  // issue one 64x32 X-tile + 64x32 W-tile into LDS buffer `bufi` (8 b128/lane)
  auto issue = [&](int k0, int bufi) {
#pragma unroll
    for (int i = 0; i < 4; ++i) {                 // 512 float4 / 128 threads
      int f = tid + 128 * i;
      int row = f >> 3, c4 = (f & 7) << 2;
      async_ld_b128(&X[(size_t)(m0 + row) * K + k0 + c4], &lx[bufi][row * 36 + c4]);
      async_ld_b128(&W[(size_t)(n0 + row) * K + k0 + c4], &lw[bufi][row * 36 + c4]);
    }
  };

  const int NK = K >> 5;
  issue(0, 0);

  for (int kk = 0; kk < NK; ++kk) {
    const int cur = kk & 1;
    if (kk + 1 < NK) {
      issue((kk + 1) << 5, cur ^ 1);
      asm volatile("s_wait_asynccnt 0x8" ::: "memory");   // current 8 loads done
    } else {
      asm volatile("s_wait_asynccnt 0x0" ::: "memory");
    }
    __syncthreads();

    // A fragment: wave's 16 rows, f32 -> f16
    const int mrow = wid * 16 + l15;
    const int kb = half ? 8 : 0;
    f4 x0 = *(const f4*)&lx[cur][mrow * 36 + kb];
    f4 x1 = *(const f4*)&lx[cur][mrow * 36 + kb + 4];
    f4 x2 = *(const f4*)&lx[cur][mrow * 36 + 16 + kb];
    f4 x3 = *(const f4*)&lx[cur][mrow * 36 + 16 + kb + 4];
    v16h a;
#pragma unroll
    for (int i = 0; i < 4; ++i) {
      a[i]      = (_Float16)x0[i];
      a[4 + i]  = (_Float16)x1[i];
      a[8 + i]  = (_Float16)x2[i];
      a[12 + i] = (_Float16)x3[i];
    }

    const int kb2 = half ? 8 : 0;
#pragma unroll
    for (int t = 0; t < 4; ++t) {
      const int nrow = t * 16 + l15;
      f4 b0 = *(const f4*)&lw[cur][nrow * 36 + kb2];
      f4 b1 = *(const f4*)&lw[cur][nrow * 36 + kb2 + 4];
      f4 b2 = *(const f4*)&lw[cur][nrow * 36 + kb2 + 16];
      f4 b3 = *(const f4*)&lw[cur][nrow * 36 + kb2 + 20];
      v16h bb;
#pragma unroll
      for (int i = 0; i < 4; ++i) {
        bb[i]      = (_Float16)b0[i];
        bb[4 + i]  = (_Float16)b1[i];
        bb[8 + i]  = (_Float16)b2[i];
        bb[12 + i] = (_Float16)b3[i];
      }
      acc[t] = __builtin_amdgcn_wmma_f32_16x16x32_f16(false, a, false, bb,
                                                      (short)0, acc[t], false, false);
    }
    __syncthreads();   // protect the buffer the next iteration's issue overwrites
  }

  // Epilogue: + bias, store. D layout: lane n=l15, VGPR r -> m = r + 8*half.
#pragma unroll
  for (int t = 0; t < 4; ++t) {
    const int n = n0 + t * 16 + l15;
    const float bv = bias[n];
#pragma unroll
    for (int r = 0; r < 8; ++r) {
      const int m = m0 + wid * 16 + r + 8 * half;
      float val = acc[t][r] + bv;
      if (TRANS) {
        const int bb = m >> 10;          // S_ == 1024
        const int q  = m & (S_ - 1);
        Y[(size_t)bb * E_ * S_ + (size_t)n * S_ + q] = (TOUT)val;
      } else {
        Y[(size_t)m * N + n] = (TOUT)val;
      }
    }
  }
}

// ---------------------------------------------------------------------------
// Fused flash attention + head-mean attn weights.
// grid = (S/64, H, B); block = 128 (4 waves x 16 queries).
// Q,K: f16 [B*S, E]; Vt: f16 [B][E][S]; ctx: f32 [B,S,E]; attnw: f32 [B,S,S].
// ---------------------------------------------------------------------------
__global__ __launch_bounds__(128)
void attn_fused(const _Float16* __restrict__ Q, const _Float16* __restrict__ Kh,
                const _Float16* __restrict__ Vt, const unsigned char* __restrict__ pad,
                float* __restrict__ ctx, float* __restrict__ attnw)
{
  __shared__ _Float16 ldsP[4][16][32];      // per-wave P relayout tile
  const int tid = threadIdx.x, wid = tid >> 5, lane = tid & 31;
  const int half = lane >> 4, l15 = lane & 15;
  const int qw = blockIdx.x * 64 + wid * 16;
  const int h = blockIdx.y, b = blockIdx.z;
  const float scale = 0.125f;               // 1/sqrt(Dh)

  // Q A-fragments (Dh=64 -> two K=32 halves), loaded once
  const _Float16* qrow = Q + ((size_t)(b * S_ + qw + l15)) * E_ + h * DH_;
  const int kbA = half ? 8 : 0;
  const v16h aq0 = load_afrag(qrow, kbA);
  const v16h aq1 = load_afrag(qrow + 32, kbA);

  const unsigned char* padb = pad + (size_t)b * S_;
  const _Float16* Kb = Kh + ((size_t)(b * S_)) * E_ + h * DH_;
  const _Float16* Vb = Vt + (size_t)b * E_ * S_ + (size_t)h * DH_ * S_;
  const int kb2 = half ? 8 : 0;

  float mrun[8], lrun[8];
  v8f cacc[4] = {};
#pragma unroll
  for (int r = 0; r < 8; ++r) { mrun[r] = -1e30f; lrun[r] = 0.f; }

  const int nkt = ((qw + 15) >> 5) + 1;     // causal: 32-key steps

  for (int kt = 0; kt < nkt; ++kt) {
    const int key0 = kt * 32;
    // --- scores: two 16-key tiles, each Dh split in two K=32 WMMAs ---
    v8f s0 = {}, s1 = {};
    {
      const _Float16* kr0 = Kb + (size_t)(key0 + l15) * E_ + kb2;
      const _Float16* kr1 = Kb + (size_t)(key0 + 16 + l15) * E_ + kb2;
      v16h bk = load_bfrag(kr0);
      s0 = __builtin_amdgcn_wmma_f32_16x16x32_f16(false, aq0, false, bk, (short)0, s0, false, false);
      bk = load_bfrag(kr0 + 32);
      s0 = __builtin_amdgcn_wmma_f32_16x16x32_f16(false, aq1, false, bk, (short)0, s0, false, false);
      bk = load_bfrag(kr1);
      s1 = __builtin_amdgcn_wmma_f32_16x16x32_f16(false, aq0, false, bk, (short)0, s1, false, false);
      bk = load_bfrag(kr1 + 32);
      s1 = __builtin_amdgcn_wmma_f32_16x16x32_f16(false, aq1, false, bk, (short)0, s1, false, false);
    }

    // --- mask + online softmax (row stats live in the D-layout lane groups) ---
    const int kA = key0 + l15, kB = key0 + 16 + l15;
    const bool mA = padb[kA] != 0, mB = padb[kB] != 0;
    float alpha[8];
#pragma unroll
    for (int r = 0; r < 8; ++r) {
      const int q = qw + r + 8 * half;
      float v0 = s0[r] * scale; if (mA || kA > q) v0 = -1e30f;
      float v1 = s1[r] * scale; if (mB || kB > q) v1 = -1e30f;
      float tm = fmaxf(v0, v1);
#pragma unroll
      for (int sft = 1; sft < 16; sft <<= 1) tm = fmaxf(tm, __shfl_xor(tm, sft, 32));
      const float mnew = fmaxf(mrun[r], tm);
      alpha[r] = __expf(mrun[r] - mnew);
      mrun[r] = mnew;
      v0 = __expf(v0 - mnew);
      v1 = __expf(v1 - mnew);
      float rs = v0 + v1;
#pragma unroll
      for (int sft = 1; sft < 16; sft <<= 1) rs += __shfl_xor(rs, sft, 32);
      lrun[r] = lrun[r] * alpha[r] + rs;
      ldsP[wid][r + 8 * half][l15]      = (_Float16)v0;   // D-layout -> LDS
      ldsP[wid][r + 8 * half][l15 + 16] = (_Float16)v1;
    }

    // --- rescale ctx, P@V ---
#pragma unroll
    for (int t = 0; t < 4; ++t)
#pragma unroll
      for (int r = 0; r < 8; ++r) cacc[t][r] *= alpha[r];

    const v16h pa = load_afrag(&ldsP[wid][l15][0], kbA);  // same-wave DS, in-order
#pragma unroll
    for (int t = 0; t < 4; ++t) {
      const _Float16* vrow = Vb + (size_t)(t * 16 + l15) * S_ + key0 + kb2;
      const v16h bv = load_bfrag(vrow);
      cacc[t] = __builtin_amdgcn_wmma_f32_16x16x32_f16(false, pa, false, bv,
                                                       (short)0, cacc[t], false, false);
    }
  }

  // --- finalize ctx ---
  float invl[8];
#pragma unroll
  for (int r = 0; r < 8; ++r) invl[r] = 1.0f / lrun[r];
#pragma unroll
  for (int t = 0; t < 4; ++t)
#pragma unroll
    for (int r = 0; r < 8; ++r) {
      const int q = qw + r + 8 * half;
      ctx[(size_t)(b * S_ + q) * E_ + h * DH_ + t * 16 + l15] = cacc[t][r] * invl[r];
    }

  // --- pass 2: head-averaged attention weights (scores recomputed, p/H added) ---
  const float ih = 1.0f / (float)H_;
  for (int kt = 0; kt < nkt; ++kt) {
    const int key0 = kt * 32;
    v8f s0 = {}, s1 = {};
    const _Float16* kr0 = Kb + (size_t)(key0 + l15) * E_ + kb2;
    const _Float16* kr1 = Kb + (size_t)(key0 + 16 + l15) * E_ + kb2;
    v16h bk = load_bfrag(kr0);
    s0 = __builtin_amdgcn_wmma_f32_16x16x32_f16(false, aq0, false, bk, (short)0, s0, false, false);
    bk = load_bfrag(kr0 + 32);
    s0 = __builtin_amdgcn_wmma_f32_16x16x32_f16(false, aq1, false, bk, (short)0, s0, false, false);
    bk = load_bfrag(kr1);
    s1 = __builtin_amdgcn_wmma_f32_16x16x32_f16(false, aq0, false, bk, (short)0, s1, false, false);
    bk = load_bfrag(kr1 + 32);
    s1 = __builtin_amdgcn_wmma_f32_16x16x32_f16(false, aq1, false, bk, (short)0, s1, false, false);

    const int kA = key0 + l15, kB = key0 + 16 + l15;
    const bool mA = padb[kA] != 0, mB = padb[kB] != 0;
#pragma unroll
    for (int r = 0; r < 8; ++r) {
      const int q = qw + r + 8 * half;
      float* arow = attnw + (size_t)b * S_ * S_ + (size_t)q * S_;
      if (!mA && kA <= q) atomicAdd(&arow[kA], __expf(s0[r] * scale - mrun[r]) * invl[r] * ih);
      if (!mB && kB <= q) atomicAdd(&arow[kB], __expf(s1[r] * scale - mrun[r]) * invl[r] * ih);
    }
  }
}

__global__ void zero_f4(float* __restrict__ p, int n4) {
  int i = blockIdx.x * blockDim.x + threadIdx.x;
  if (i < n4) ((f4*)p)[i] = (f4){};
}

extern "C" void kernel_launch(void* const* d_in, const int* in_sizes, int n_in,
                              void* d_out, int out_size, void* d_ws, size_t ws_size,
                              hipStream_t stream) {
  const float* qin = (const float*)d_in[0];
  const float* kin = (const float*)d_in[1];
  const float* vin = (const float*)d_in[2];
  const unsigned char* mask = (const unsigned char*)d_in[3];  // numpy bool: 1B/elem
  const float* wq = (const float*)d_in[4];
  const float* wk = (const float*)d_in[5];
  const float* wv = (const float*)d_in[6];
  const float* bq = (const float*)d_in[7];
  const float* bk = (const float*)d_in[8];
  const float* bv = (const float*)d_in[9];
  const float* wo = (const float*)d_in[10];
  const float* bo = (const float*)d_in[11];

  char* ws = (char*)d_ws;
  _Float16* Qh = (_Float16*)(ws);
  _Float16* Kh = (_Float16*)(ws + ((size_t)16 << 20));
  _Float16* Vt = (_Float16*)(ws + ((size_t)32 << 20));
  float*    ctx = (float*)(ws + ((size_t)48 << 20));

  float* out   = (float*)d_out;                 // [B,S,E]
  float* attnw = out + (size_t)M_ * E_;         // [B,S,S]

  // attn weights are accumulated atomically -> must be zeroed every call
  {
    const int n4 = (B_ * S_ * S_) / 4;
    zero_f4<<<(n4 + 255) / 256, 256, 0, stream>>>(attnw, n4);
  }

  dim3 gg(E_ / 64, M_ / 64), bb(128);
  gemm_nt_bias<_Float16, false><<<gg, bb, 0, stream>>>(qin, wq, bq, Qh, M_, E_, E_);
  gemm_nt_bias<_Float16, false><<<gg, bb, 0, stream>>>(kin, wk, bk, Kh, M_, E_, E_);
  gemm_nt_bias<_Float16, true ><<<gg, bb, 0, stream>>>(vin, wv, bv, Vt, M_, E_, E_);

  attn_fused<<<dim3(S_ / 64, H_, B_), 128, 0, stream>>>(Qh, Kh, Vt, mask, ctx, attnw);

  gemm_nt_bias<float, false><<<gg, bb, 0, stream>>>(ctx, wo, bo, out, M_, E_, E_);
}
